// DynamicTextures_2808908612175
// MI455X (gfx1250) — matrix operand order repaired and checked by reading, not compiled, not run on hardware
//
#include <hip/hip_runtime.h>

// ---------------------------------------------------------------------------
// DynamicTextures for MI455X (gfx1250, wave32, WMMA + TDM)
//   MLP (f32 VALU, weights written pre-swizzled to WMMA A-fragment order)
//   -> dynamic conv1 + conv2 as implicit GEMM with v_wmma_f32_16x16x32_bf16,
//      per-sample filters staged into LDS via tensor_load_to_lds (TDM).
// ---------------------------------------------------------------------------

typedef __attribute__((ext_vector_type(16))) __bf16        bf16x16;
typedef __attribute__((ext_vector_type(8)))  float         f32x8;
typedef __attribute__((ext_vector_type(8)))  unsigned int  u32x8;
typedef __attribute__((ext_vector_type(4)))  unsigned int  u32x4;
typedef __attribute__((ext_vector_type(8)))  int           i32x8;
typedef __attribute__((ext_vector_type(4)))  int           i32x4;

#define B_SZ    16
#define D_OUT   1024
#define D_HID   512
#define D_TEX   16
#define D_INTER 32
#define TEXD    256
#define N0      9216      // 32*32*9
#define N1      4608      // 16*32*9
#define D_FLT   13824

#define XT      128       // x span per conv block
#define XL      (XT + 2)  // staged columns incl. halo
#define CSTRIDE 34        // padded per-x channel stride in LDS (ushort units)

__device__ __forceinline__ unsigned short f2bf(float f) {
  unsigned u = __builtin_bit_cast(unsigned, f);
  u += 0x7FFFu + ((u >> 16) & 1u);            // round-to-nearest-even
  return (unsigned short)(u >> 16);
}
__device__ __forceinline__ float lrelu(float v) { return v >= 0.f ? v : 0.2f * v; }

// ---------------------------------------------------------------------------
// TDM: 1-row tensor DMA of n_u16 bf16 elements, global -> LDS.
// D# per CDNA5 ISA: group0 {count=1, lds_addr,
// global_addr[56:0], type=2}; group1 {data_size=2B, tensor_dim0=tile_dim0=n}.
// 6-arg builtin form (amdgpu-toolchain / clang-23).
// ---------------------------------------------------------------------------
__device__ __forceinline__ void tdm_load_lds(unsigned lds_off, const void* gsrc,
                                             unsigned n_u16) {
  unsigned long long ga = (unsigned long long)(uintptr_t)gsrc;
  u32x4 g0;
  g0[0] = 1u;                                              // count=1 (user D#)
  g0[1] = lds_off;                                         // lds_addr
  g0[2] = (unsigned)(ga & 0xFFFFFFFFu);                    // global_addr[31:0]
  g0[3] = (unsigned)((ga >> 32) & 0x01FFFFFFu) | 0x80000000u; // [56:32] | type=2
  i32x8 g1;
  g1[0] = (int)(1u << 16);                                 // data_size=1 -> 2B
  g1[1] = (int)((n_u16 & 0xFFFFu) << 16);                  // tensor_dim0 lo16
  g1[2] = (int)((n_u16 >> 16) | (1u << 16));               // dim0 hi16, dim1=1
  g1[3] = (int)((n_u16 & 0xFFFFu) << 16);                  // tile_dim0
  g1[4] = 1;                                               // tile_dim1=1
  g1[5] = (int)n_u16;                                      // dim0_stride lo32
  g1[6] = 0;
  g1[7] = 0;
  i32x4 z4 = {0, 0, 0, 0};
  i32x8 z8 = {0, 0, 0, 0, 0, 0, 0, 0};
  __builtin_amdgcn_tensor_load_to_lds(g0, g1, z4, z4, z8, 0);
}

// ---------------------------------------------------------------------------
// MLP layer: out[b][j] = lrelu(sum_k in[b][k] * W[j][k] + bias[j])
// ---------------------------------------------------------------------------
__global__ __launch_bounds__(256)
void mlp_layer_kernel(const float* __restrict__ in, const float* __restrict__ W,
                      const float* __restrict__ bias, float* __restrict__ out,
                      int din) {
  const int j = blockIdx.x * 256 + threadIdx.x;          // 0..511
  float acc[B_SZ];
  const float bv = bias[j];
#pragma unroll
  for (int b = 0; b < B_SZ; ++b) acc[b] = bv;
  const float* wr = W + (size_t)j * din;
  for (int k = 0; k < din; ++k) {
    const float wv = wr[k];
#pragma unroll
    for (int b = 0; b < B_SZ; ++b) acc[b] = fmaf(wv, in[b * din + k], acc[b]);
  }
#pragma unroll
  for (int b = 0; b < B_SZ; ++b) out[b * D_HID + j] = lrelu(acc[b]);
}

// ---------------------------------------------------------------------------
// MLP head: weights[b][j] = sum_k h[b][k]*W3[j][k] + b3[j], stored bf16 and
// PRE-SWIZZLED to WMMA A-fragment order: [tap][co][ci] (ci contiguous).
// ---------------------------------------------------------------------------
__global__ __launch_bounds__(256)
void mlp_head_kernel(const float* __restrict__ h, const float* __restrict__ W3,
                     const float* __restrict__ b3,
                     unsigned short* __restrict__ wq1,   // [16][9][32][32]
                     unsigned short* __restrict__ wq2) { // [16][9][16][32]
  const int j = blockIdx.x * 256 + threadIdx.x;          // 0..13823
  float acc[B_SZ];
  const float bv = b3[j];
#pragma unroll
  for (int b = 0; b < B_SZ; ++b) acc[b] = bv;
  const float* wr = W3 + (size_t)j * D_HID;
  for (int k = 0; k < D_HID; ++k) {
    const float wv = wr[k];
#pragma unroll
    for (int b = 0; b < B_SZ; ++b) acc[b] = fmaf(wv, h[b * D_HID + k], acc[b]);
  }
  if (j < N0) {                       // conv1 filter: j = co*288 + ci*9 + t
    const int co = j / 288, ci = (j % 288) / 9, t = j % 9;
    const int dst = (t * D_INTER + co) * D_INTER + ci;
#pragma unroll
    for (int b = 0; b < B_SZ; ++b) wq1[(size_t)b * N0 + dst] = f2bf(acc[b]);
  } else {                            // conv2 filter: co<16
    const int jj = j - N0;
    const int co = jj / 288, ci = (jj % 288) / 9, t = jj % 9;
    const int dst = (t * D_TEX + co) * D_INTER + ci;
#pragma unroll
    for (int b = 0; b < B_SZ; ++b) wq2[(size_t)b * N1 + dst] = f2bf(acc[b]);
  }
}

// ---------------------------------------------------------------------------
// conv1: x1[b][co][y][x] = bf16(lrelu( sum_{ci,ky,kx} w0 * tex )), 32 out ch
// Implicit GEMM: M=co(16x2), N=x(16x8), K=288 (9 taps x 32 ci, 1 WMMA/tap)
// ---------------------------------------------------------------------------
__global__ __launch_bounds__(256)
void conv1_kernel(const float* __restrict__ tex, const unsigned short* __restrict__ wq1,
                  unsigned short* __restrict__ x1) {
  __shared__ unsigned short s_w[N0];                 // [t][co][ci] bf16
  __shared__ unsigned short s_in[3 * XL * CSTRIDE];  // [row][x][ci] bf16

  const int tid = threadIdx.x;
  const int x0  = blockIdx.x * XT;   // 0 or 128
  const int y   = blockIdx.y;        // 0..255
  const int b   = blockIdx.z;        // 0..15

  // kick per-sample filter DMA (wave 0), overlap with input staging below
  if (tid < 32) {
    tdm_load_lds((unsigned)(uintptr_t)&s_w[0], wq1 + (size_t)b * N0, N0);
  }

  // stage input rows y-1..y+1, x0-1..x0+128 transposed to [row][x][ci];
  // two input channels per iteration -> packed b32 LDS stores
  unsigned int* s_in32 = (unsigned int*)s_in;
  for (int i = tid; i < 3 * XL * (D_INTER / 2); i += 256) {
    const int x   = i % XL;
    const int cp  = (i / XL) % (D_INTER / 2);
    const int row = i / (XL * (D_INTER / 2));
    const int gy  = y + row - 1;
    const int gx  = x0 - 1 + x;
    float v0 = 0.f, v1 = 0.f;
    if ((unsigned)gy < TEXD && (unsigned)gx < TEXD) {
      v0 = tex[((size_t)(2 * cp) * TEXD + gy) * TEXD + gx];
      v1 = tex[((size_t)(2 * cp + 1) * TEXD + gy) * TEXD + gx];
    }
    s_in32[(row * XL + x) * (CSTRIDE / 2) + cp] =
        (unsigned)f2bf(v0) | ((unsigned)f2bf(v1) << 16);
  }
  if (tid < 32) __builtin_amdgcn_s_wait_tensorcnt(0);
  __syncthreads();

  const int wave = tid >> 5, lane = tid & 31;
  const int n = lane & 15, g = lane >> 4;
  const int c = wave & 1;            // co chunk (0..1)
  const int tpair = wave >> 1;       // 0..3 -> two x-tiles each

  // A fragments from swizzled filters: two b128 LDS loads per fragment
  const unsigned int* s_w32 = (const unsigned int*)s_w;
  bf16x16 afr[9];
#pragma unroll
  for (int t = 0; t < 9; ++t) {
    const int basew = (t * D_INTER + c * 16 + n) * 16 + g * 4;
    u32x8 au;
#pragma unroll
    for (int v = 0; v < 8; ++v) au[v] = s_w32[basew + (v >> 2) * 8 + (v & 3)];
    afr[t] = __builtin_bit_cast(bf16x16, au);
  }

#pragma unroll
  for (int ti = 0; ti < 2; ++ti) {
    const int xt = tpair * 2 + ti;
    f32x8 acc = {0.f, 0.f, 0.f, 0.f, 0.f, 0.f, 0.f, 0.f};
#pragma unroll
    for (int t = 0; t < 9; ++t) {
      const int dy = t / 3, dx = t % 3;
      const int xloc = xt * 16 + n + dx;                  // halo-adjusted
      const int base32 = (dy * XL + xloc) * (CSTRIDE / 2) + g * 4;
      u32x8 bu;
#pragma unroll
      for (int v = 0; v < 8; ++v) bu[v] = s_in32[base32 + (v >> 2) * 8 + (v & 3)];
      const bf16x16 bfr = __builtin_bit_cast(bf16x16, bu);
      acc = __builtin_amdgcn_wmma_f32_16x16x32_bf16(false, afr[t], false, bfr,
                                                    (short)0, acc, false, false);
    }
    const int x = x0 + xt * 16 + n;
#pragma unroll
    for (int r = 0; r < 8; ++r) {
      const int co = c * 16 + g * 8 + r;
      x1[(((size_t)b * D_INTER + co) * TEXD + y) * TEXD + x] = f2bf(lrelu(acc[r]));
    }
  }
}

// ---------------------------------------------------------------------------
// conv2: out[b][co][y][x] = sum_{ci,ky,kx} w1 * x1   (16 out ch, f32 out)
// ---------------------------------------------------------------------------
__global__ __launch_bounds__(256)
void conv2_kernel(const unsigned short* __restrict__ x1, const unsigned short* __restrict__ wq2,
                  float* __restrict__ out) {
  __shared__ unsigned short s_w[N1];                 // [t][co][ci] bf16
  __shared__ unsigned short s_in[3 * XL * CSTRIDE];

  const int tid = threadIdx.x;
  const int x0  = blockIdx.x * XT;
  const int y   = blockIdx.y;
  const int b   = blockIdx.z;

  if (tid < 32) {
    tdm_load_lds((unsigned)(uintptr_t)&s_w[0], wq2 + (size_t)b * N1, N1);
  }

  unsigned int* s_in32 = (unsigned int*)s_in;
  for (int i = tid; i < 3 * XL * (D_INTER / 2); i += 256) {
    const int x   = i % XL;
    const int cp  = (i / XL) % (D_INTER / 2);
    const int row = i / (XL * (D_INTER / 2));
    const int gy  = y + row - 1;
    const int gx  = x0 - 1 + x;
    unsigned v0 = 0, v1 = 0;
    if ((unsigned)gy < TEXD && (unsigned)gx < TEXD) {
      v0 = x1[(((size_t)b * D_INTER + 2 * cp) * TEXD + gy) * TEXD + gx];
      v1 = x1[(((size_t)b * D_INTER + 2 * cp + 1) * TEXD + gy) * TEXD + gx];
    }
    s_in32[(row * XL + x) * (CSTRIDE / 2) + cp] = v0 | (v1 << 16);
  }
  if (tid < 32) __builtin_amdgcn_s_wait_tensorcnt(0);
  __syncthreads();

  const int wave = tid >> 5, lane = tid & 31;
  const int n = lane & 15, g = lane >> 4;
  const int xt = wave;               // one 16-px tile per wave

  const unsigned int* s_w32 = (const unsigned int*)s_w;
  bf16x16 afr[9];
#pragma unroll
  for (int t = 0; t < 9; ++t) {
    const int basew = (t * D_TEX + n) * 16 + g * 4;
    u32x8 au;
#pragma unroll
    for (int v = 0; v < 8; ++v) au[v] = s_w32[basew + (v >> 2) * 8 + (v & 3)];
    afr[t] = __builtin_bit_cast(bf16x16, au);
  }

  f32x8 acc = {0.f, 0.f, 0.f, 0.f, 0.f, 0.f, 0.f, 0.f};
#pragma unroll
  for (int t = 0; t < 9; ++t) {
    const int dy = t / 3, dx = t % 3;
    const int xloc = xt * 16 + n + dx;
    const int base32 = (dy * XL + xloc) * (CSTRIDE / 2) + g * 4;
    u32x8 bu;
#pragma unroll
    for (int v = 0; v < 8; ++v) bu[v] = s_in32[base32 + (v >> 2) * 8 + (v & 3)];
    const bf16x16 bfr = __builtin_bit_cast(bf16x16, bu);
    acc = __builtin_amdgcn_wmma_f32_16x16x32_bf16(false, afr[t], false, bfr,
                                                  (short)0, acc, false, false);
  }
  const int x = x0 + xt * 16 + n;
#pragma unroll
  for (int r = 0; r < 8; ++r) {
    const int co = g * 8 + r;
    out[(((size_t)b * D_TEX + co) * TEXD + y) * TEXD + x] = acc[r];
  }
}

// ---------------------------------------------------------------------------
extern "C" void kernel_launch(void* const* d_in, const int* in_sizes, int n_in,
                              void* d_out, int out_size, void* d_ws, size_t ws_size,
                              hipStream_t stream) {
  const float* z_face = (const float*)d_in[0];
  const float* tex    = (const float*)d_in[1];
  const float* W1     = (const float*)d_in[2];
  const float* b1     = (const float*)d_in[3];
  const float* W2     = (const float*)d_in[4];
  const float* b2     = (const float*)d_in[5];
  const float* W3     = (const float*)d_in[6];
  const float* b3     = (const float*)d_in[7];
  float* out = (float*)d_out;

  char* ws = (char*)d_ws;
  float*          h1  = (float*)(ws);                    // 16*512 f32  = 32 KB
  float*          h2  = (float*)(ws + 32768);            // 16*512 f32  = 32 KB
  unsigned short* wq1 = (unsigned short*)(ws + 65536);   // 16*9216 bf16 = 288 KB
  unsigned short* wq2 = (unsigned short*)(ws + 360448);  // 16*4608 bf16 = 144 KB
  unsigned short* x1  = (unsigned short*)(ws + 524288);  // 16*32*256*256 bf16 = 64 MB

  mlp_layer_kernel<<<dim3(2), dim3(256), 0, stream>>>(z_face, W1, b1, h1, D_OUT);
  mlp_layer_kernel<<<dim3(2), dim3(256), 0, stream>>>(h1, W2, b2, h2, D_HID);
  mlp_head_kernel<<<dim3(54), dim3(256), 0, stream>>>(h2, W3, b3, wq1, wq2);

  conv1_kernel<<<dim3(2, 256, 16), dim3(256), 0, stream>>>(tex, wq1, x1);
  conv2_kernel<<<dim3(2, 256, 16), dim3(256), 0, stream>>>(x1, wq2, out);
}
